// GlobalSphereAttention_70368744178019
// MI455X (gfx1250) — matrix operand order repaired and checked by reading, not compiled, not run on hardware
//
#include <hip/hip_runtime.h>
#include <hip/hip_bf16.h>

#define DIMX 512
#define HEADS 8
#define DH 64
#define INNER 512
#define BATCH 2
#define SEQ 2048
#define BN (BATCH * SEQ)   // 4096
#define EPSLN 1e-5f
#define SCALE 0.125f       // DH^-0.5

typedef __attribute__((ext_vector_type(16))) __bf16 v16bf;
typedef __attribute__((ext_vector_type(8)))  float  v8f;

// A-fragment K-index map for 16-bit 16x16x32 WMMA:
// lanes 0-15 hold K {0..15}, lanes 16-31 hold K {16..31}
static __device__ __forceinline__ int kmap(int e, int half) {
    return e + 16 * half;
}

// ---------------- weight prep: transpose + f32->bf16 ----------------
__global__ void prep_wqkv(const float* __restrict__ w, __bf16* __restrict__ wT) {
    int idx = blockIdx.x * 256 + threadIdx.x;       // 1536*512 elements
    int r = idx >> 9;                               // output row = original col
    int c = idx & 511;                              // k
    wT[idx] = (__bf16)w[(size_t)c * (3 * INNER) + r];
}

__global__ void prep_wout(const float* __restrict__ w, __bf16* __restrict__ wT) {
    int idx = blockIdx.x * 256 + threadIdx.x;       // 512*512 elements
    int r = idx >> 9;
    int c = idx & 511;
    wT[idx] = (__bf16)w[(size_t)c * DIMX + r];
}

// ---------------- LayerNorm -> bf16 ----------------
__global__ void __launch_bounds__(256)
ln_kernel(const float* __restrict__ x, const float* __restrict__ g,
          const float* __restrict__ bsh, __bf16* __restrict__ xn) {
    int wave = threadIdx.x >> 5;
    int lane = threadIdx.x & 31;
    int row  = blockIdx.x * 8 + wave;
    const float* xr = x + (size_t)row * DIMX;
    float v[16];
    float s = 0.f;
#pragma unroll
    for (int i = 0; i < 16; ++i) { v[i] = xr[lane + 32 * i]; s += v[i]; }
#pragma unroll
    for (int o = 16; o >= 1; o >>= 1) s += __shfl_xor(s, o, 32);
    float mu = s * (1.0f / DIMX);
    float vs = 0.f;
#pragma unroll
    for (int i = 0; i < 16; ++i) { float d = v[i] - mu; vs += d * d; }
#pragma unroll
    for (int o = 16; o >= 1; o >>= 1) vs += __shfl_xor(vs, o, 32);
    float rinv = rsqrtf(vs * (1.0f / DIMX) + EPSLN);
    __bf16* outr = xn + (size_t)row * DIMX;
#pragma unroll
    for (int i = 0; i < 16; ++i) {
        int c = lane + 32 * i;
        outr[c] = (__bf16)(((v[i] - mu) * rinv) * g[c] + bsh[c]);
    }
}

// ---------------- QKV GEMM: xn[4096x512] @ wqkv[512x1536] ----------------
// Each wave computes a 16x64 output tile; a whole 64-deep K slab of loads
// (2 A frags + 8 B frags) is pinned before the WMMAs via sched_barrier so the
// backend keeps them in-flight (partial s_wait_loadcnt) instead of wait-0.
__global__ void __launch_bounds__(256)
qkv_gemm(const __bf16* __restrict__ xn, const __bf16* __restrict__ wT,
         __bf16* __restrict__ q, __bf16* __restrict__ k, __bf16* __restrict__ vt) {
    int wave = threadIdx.x >> 5;
    int lane = threadIdx.x & 31;
    int half = lane >> 4;
    int l15  = lane & 15;
    int rowbase = blockIdx.x * 16;
    int col0    = blockIdx.y * 512 + wave * 64;
    v8f acc[4] = {};
    for (int kb = 0; kb < DIMX; kb += 64) {
        v16bf a[2], bfr[2][4];
#pragma unroll
        for (int u = 0; u < 2; ++u) {
            const __bf16* ar = xn + (size_t)(rowbase + l15) * DIMX + kb + 32 * u;
#pragma unroll
            for (int e = 0; e < 16; ++e) a[u][e] = ar[kmap(e, half)];
#pragma unroll
            for (int t = 0; t < 4; ++t)
                bfr[u][t] = *(const v16bf*)(wT + (size_t)(col0 + 16 * t + l15) * DIMX
                                            + kb + 32 * u + 16 * half);
        }
        __builtin_amdgcn_sched_barrier(0);   // keep slab loads ahead of the WMMAs
#pragma unroll
        for (int u = 0; u < 2; ++u)
#pragma unroll
            for (int t = 0; t < 4; ++t)
                acc[t] = __builtin_amdgcn_wmma_f32_16x16x32_bf16(false, a[u], false, bfr[u][t],
                                                                 (short)0, acc[t], false, false);
    }
#pragma unroll
    for (int t = 0; t < 4; ++t) {
        int col = col0 + 16 * t + l15;
#pragma unroll
        for (int r = 0; r < 8; ++r) {
            int grow = rowbase + r + 8 * half;
            int b_ = grow >> 11;            // / SEQ
            int n  = grow & (SEQ - 1);
            float v = acc[t][r];
            if (col < INNER) {
                int h = col >> 6, d = col & 63;
                q[((size_t)(b_ * HEADS + h) * SEQ + n) * DH + d] = (__bf16)(v * SCALE);
            } else if (col < 2 * INNER) {
                int c2 = col - INNER; int h = c2 >> 6, d = c2 & 63;
                k[((size_t)(b_ * HEADS + h) * SEQ + n) * DH + d] = (__bf16)v;
            } else {
                int c2 = col - 2 * INNER; int h = c2 >> 6, d = c2 & 63;
                vt[((size_t)(b_ * HEADS + h) * DH + d) * SEQ + n] = (__bf16)v;
            }
        }
    }
}

// ---------------- fused attention with cross-head normalization ----------------
// block = 256 threads = 8 waves = 8 heads; each block owns 16 query rows of one batch.
__global__ void __launch_bounds__(256, 1)
attn_kernel(const __bf16* __restrict__ q, const __bf16* __restrict__ kmat,
            const __bf16* __restrict__ vt, const float* __restrict__ sph,
            float* __restrict__ attn_out) {
    __shared__ float sS[HEADS][16][32];   // 16 KB score-exchange tile
    int h    = threadIdx.x >> 5;
    int lane = threadIdx.x & 31;
    int half = lane >> 4;
    int l15  = lane & 15;
    int b    = blockIdx.x >> 7;           // 128 i-tiles per batch
    int i0   = (blockIdx.x & 127) * 16;
    size_t bh = (size_t)(b * HEADS + h);

    // Q fragments (16x64, scale pre-folded), A-layout
    v16bf qa0, qa1;
    {
        const __bf16* qr = q + (bh * SEQ + i0 + l15) * DH;
#pragma unroll
        for (int e = 0; e < 16; ++e) {
            qa0[e] = qr[kmap(e, half)];
            qa1[e] = qr[32 + kmap(e, half)];
        }
    }
    v8f o0 = {}, o1 = {}, o2 = {}, o3 = {};
    float mrow = -1e30f, lrow = 0.f;

    const __bf16* kbase  = kmat + bh * SEQ * DH;
    const __bf16* vtbase = vt + bh * DH * SEQ;

    for (int j0 = 0; j0 < SEQ; j0 += 32) {
        __syncthreads();                  // sS reusable
        // prefetch next j-tile's K rows and Vt rows (gfx1250 global_prefetch_b8)
        int jn = j0 + 32;
        if (jn < SEQ) {
            __builtin_prefetch(kbase + (size_t)(jn + lane) * DH, 0, 0);
            __builtin_prefetch(vtbase + (size_t)lane * SEQ + jn, 0, 0);
            __builtin_prefetch(vtbase + (size_t)(32 + lane) * SEQ + jn, 0, 0);
        }
        // S = Q K^T : two 16x16 C tiles (keys j0..+15, j0+16..+31)
        v8f c0 = {}, c1 = {};
        {
            const __bf16* kb = kbase + (size_t)j0 * DH;
            v16bf kb00 = *(const v16bf*)(kb + (size_t)l15 * DH + 16 * half);
            v16bf kb10 = *(const v16bf*)(kb + (size_t)(l15 + 16) * DH + 16 * half);
            v16bf kb01 = *(const v16bf*)(kb + (size_t)l15 * DH + 32 + 16 * half);
            v16bf kb11 = *(const v16bf*)(kb + (size_t)(l15 + 16) * DH + 32 + 16 * half);
            __builtin_amdgcn_sched_barrier(0);
            c0 = __builtin_amdgcn_wmma_f32_16x16x32_bf16(false, qa0, false, kb00, (short)0, c0, false, false);
            c1 = __builtin_amdgcn_wmma_f32_16x16x32_bf16(false, qa0, false, kb10, (short)0, c1, false, false);
            c0 = __builtin_amdgcn_wmma_f32_16x16x32_bf16(false, qa1, false, kb01, (short)0, c0, false, false);
            c1 = __builtin_amdgcn_wmma_f32_16x16x32_bf16(false, qa1, false, kb11, (short)0, c1, false, false);
        }
#pragma unroll
        for (int r = 0; r < 8; ++r) {
            sS[h][r + 8 * half][l15]      = c0[r];
            sS[h][r + 8 * half][l15 + 16] = c1[r];
        }
        __syncthreads();
        // cross-head norm: 256 threads x 2 elements each; apply (1+sph) then mean/ddof-1 std
#pragma unroll
        for (int rep = 0; rep < 2; ++rep) {
            int e  = threadIdx.x + rep * 256;
            int M  = e >> 5, jj = e & 31;
            float w = 1.0f + sph[(size_t)(i0 + M) * SEQ + j0 + jj];
            float vals[HEADS];
            float mean = 0.f;
#pragma unroll
            for (int hh = 0; hh < HEADS; ++hh) { vals[hh] = sS[hh][M][jj] * w; mean += vals[hh]; }
            mean *= (1.0f / HEADS);
            float ss = 0.f;
#pragma unroll
            for (int hh = 0; hh < HEADS; ++hh) { float d = vals[hh] - mean; ss += d * d; }
            float rstd = rsqrtf(ss * (1.0f / (HEADS - 1)) + 1e-30f);
#pragma unroll
            for (int hh = 0; hh < HEADS; ++hh) sS[hh][M][jj] = (vals[hh] - mean) * rstd;
        }
        __syncthreads();
        // read normalized tile back in A-fragment order; online softmax
        float p[16];
        float tmax = -1e30f;
#pragma unroll
        for (int e = 0; e < 16; ++e) { p[e] = sS[h][l15][kmap(e, half)]; tmax = fmaxf(tmax, p[e]); }
        tmax = fmaxf(tmax, __shfl_xor(tmax, 16, 32));
        float mnew  = fmaxf(mrow, tmax);
        float alpha = __expf(mrow - mnew);
        float rsum = 0.f;
#pragma unroll
        for (int e = 0; e < 16; ++e) { p[e] = __expf(p[e] - mnew); rsum += p[e]; }
        rsum += __shfl_xor(rsum, 16, 32);
        lrow = lrow * alpha + rsum;
        mrow = mnew;
        // V B-fragments from transposed V: contiguous 32B per lane (issue early)
        const __bf16* vb = vtbase + (size_t)l15 * SEQ + j0 + 16 * half;
        v16bf vb0 = *(const v16bf*)(vb + 0 * 16 * SEQ);
        v16bf vb1 = *(const v16bf*)(vb + 1 * 16 * SEQ);
        v16bf vb2 = *(const v16bf*)(vb + 2 * 16 * SEQ);
        v16bf vb3 = *(const v16bf*)(vb + 3 * 16 * SEQ);
        __builtin_amdgcn_sched_barrier(0);
        // rescale O: factor for row (r+8*half) lives in lane (r+8*half)
#pragma unroll
        for (int r = 0; r < 8; ++r) {
            float f = __shfl(alpha, r + 8 * half, 32);
            o0[r] *= f; o1[r] *= f; o2[r] *= f; o3[r] *= f;
        }
        // P fragment is already in A-lane order
        v16bf pa;
#pragma unroll
        for (int e = 0; e < 16; ++e) pa[e] = (__bf16)p[e];
        o0 = __builtin_amdgcn_wmma_f32_16x16x32_bf16(false, pa, false, vb0, (short)0, o0, false, false);
        o1 = __builtin_amdgcn_wmma_f32_16x16x32_bf16(false, pa, false, vb1, (short)0, o1, false, false);
        o2 = __builtin_amdgcn_wmma_f32_16x16x32_bf16(false, pa, false, vb2, (short)0, o2, false, false);
        o3 = __builtin_amdgcn_wmma_f32_16x16x32_bf16(false, pa, false, vb3, (short)0, o3, false, false);
    }
    // finalize: divide by softmax denom; write [B, N, H*DH] f32
    float linv = 1.0f / lrow;
#pragma unroll
    for (int r = 0; r < 8; ++r) {
        float f = __shfl(linv, r + 8 * half, 32);
        int row = i0 + r + 8 * half;
        float* orow = attn_out + ((size_t)(b * SEQ) + row) * INNER + h * DH;
        orow[l15 +  0] = o0[r] * f;
        orow[l15 + 16] = o1[r] * f;
        orow[l15 + 32] = o2[r] * f;
        orow[l15 + 48] = o3[r] * f;
    }
}

// ---------------- output projection: aout[4096x512] @ wout[512x512] + b ----------------
// Each wave computes a 16x64 output tile with the same slab-loaded pipeline.
__global__ void __launch_bounds__(256)
outproj_gemm(const float* __restrict__ ain, const __bf16* __restrict__ wT,
             const float* __restrict__ bout, float* __restrict__ out) {
    int wave = threadIdx.x >> 5;
    int lane = threadIdx.x & 31;
    int half = lane >> 4;
    int l15  = lane & 15;
    int rowbase = blockIdx.x * 16;
    int col0    = wave * 64;             // 8 waves cover all 512 columns
    v8f acc[4] = {};
    for (int kb = 0; kb < INNER; kb += 64) {
        v16bf a[2], bfr[2][4];
#pragma unroll
        for (int u = 0; u < 2; ++u) {
            const float* ar = ain + (size_t)(rowbase + l15) * INNER + kb + 32 * u;
#pragma unroll
            for (int e = 0; e < 16; ++e) a[u][e] = (__bf16)ar[kmap(e, half)];
#pragma unroll
            for (int t = 0; t < 4; ++t)
                bfr[u][t] = *(const v16bf*)(wT + (size_t)(col0 + 16 * t + l15) * INNER
                                            + kb + 32 * u + 16 * half);
        }
        __builtin_amdgcn_sched_barrier(0);   // keep slab loads ahead of the WMMAs
#pragma unroll
        for (int u = 0; u < 2; ++u)
#pragma unroll
            for (int t = 0; t < 4; ++t)
                acc[t] = __builtin_amdgcn_wmma_f32_16x16x32_bf16(false, a[u], false, bfr[u][t],
                                                                 (short)0, acc[t], false, false);
    }
#pragma unroll
    for (int t = 0; t < 4; ++t) {
        int col = col0 + 16 * t + l15;
        float bo = bout[col];
#pragma unroll
        for (int r = 0; r < 8; ++r) {
            int row = rowbase + r + 8 * half;
            out[(size_t)row * DIMX + col] = acc[t][r] + bo;
        }
    }
}

extern "C" void kernel_launch(void* const* d_in, const int* in_sizes, int n_in,
                              void* d_out, int out_size, void* d_ws, size_t ws_size,
                              hipStream_t stream) {
    const float* x     = (const float*)d_in[0];
    const float* sph   = (const float*)d_in[1];
    const float* ln_g  = (const float*)d_in[2];
    const float* ln_b  = (const float*)d_in[3];
    const float* w_qkv = (const float*)d_in[4];
    const float* w_out = (const float*)d_in[5];
    const float* b_out = (const float*)d_in[6];
    float* out = (float*)d_out;

    char* ws = (char*)d_ws;
    size_t off = 0;
    __bf16* xn    = (__bf16*)(ws + off); off += (size_t)BN * DIMX * 2;            // 4 MB
    __bf16* qw    = (__bf16*)(ws + off); off += (size_t)BATCH * INNER * SEQ * 2;  // 4 MB
    __bf16* kw    = (__bf16*)(ws + off); off += (size_t)BATCH * INNER * SEQ * 2;  // 4 MB
    __bf16* vtw   = (__bf16*)(ws + off); off += (size_t)BATCH * INNER * SEQ * 2;  // 4 MB
    float*  aout  = (float*) (ws + off); off += (size_t)BN * INNER * 4;           // 8 MB
    __bf16* wqkvT = (__bf16*)(ws + off); off += (size_t)3 * INNER * DIMX * 2;     // 1.5 MB
    __bf16* woutT = (__bf16*)(ws + off); off += (size_t)INNER * DIMX * 2;         // 0.5 MB
    (void)ws_size; (void)in_sizes; (void)n_in; (void)out_size;

    prep_wqkv<<<(3 * INNER * DIMX) / 256, 256, 0, stream>>>(w_qkv, wqkvT);
    prep_wout<<<(INNER * DIMX) / 256, 256, 0, stream>>>(w_out, woutT);
    ln_kernel<<<BN / 8, 256, 0, stream>>>(x, ln_g, ln_b, xn);
    qkv_gemm<<<dim3(BN / 16, 3), 256, 0, stream>>>(xn, wqkvT, qw, kw, vtw);
    attn_kernel<<<BATCH * (SEQ / 16), 256, 0, stream>>>(qw, kw, vtw, sph, aout);
    outproj_gemm<<<BN / 16, 256, 0, stream>>>(aout, woutT, b_out, out);
}
